// LatticeNer_22823456210979
// MI455X (gfx1250) — matrix operand order repaired and checked by reading, not compiled, not run on hardware
//
#include <hip/hip_runtime.h>
#include <math.h>

// ---------------- problem constants ----------------
#define TT 512
#define KK 6
#define GG 2048
#define DD 256
#define HH 256
#define H3 768

// ---------------- workspace layout ----------------
// f32 region (float offsets):
static const size_t OFF_CHAR  = 0;                                   // 2 * T*3H
static const size_t OFF_ALPHA = OFF_CHAR  + (size_t)2 * TT * H3;     // 2 * T*H
static const size_t OFF_WORD  = OFF_ALPHA + (size_t)2 * TT * HH;     // 2 * (G+1)*3H (row G zero)
static const size_t OFF_BUF   = OFF_WORD  + (size_t)2 * (GG + 1) * H3; // 2 * (G+1)*H
static const size_t OFF_AWT   = OFF_BUF   + (size_t)2 * (GG + 1) * HH; // 2 * H*H   (aw_hh^T)
static const size_t OFF_WHH4  = OFF_AWT   + (size_t)2 * HH * HH;     // 2 * H*H*4 (packed i,o,g,0)
static const size_t OFF_WWHH4 = OFF_WHH4  + (size_t)2 * HH * HH * 4; // 2 * H*H*4
static const size_t F32_TOTAL = OFF_WWHH4 + (size_t)2 * HH * HH * 4;
// f16 region (half offsets), base = (_Float16*)(ws + F32_TOTAL):
static const size_t HOFF_X    = 0;                                   // T*D
static const size_t HOFF_GAZ  = HOFF_X    + (size_t)TT * DD;         // G*D
static const size_t HOFF_WIH  = HOFF_GAZ  + (size_t)GG * DD;         // 2 * 3H*D
static const size_t HOFF_AWIH = HOFF_WIH  + (size_t)2 * H3 * DD;     // 2 * H*D
static const size_t HOFF_WWIH = HOFF_AWIH + (size_t)2 * HH * DD;     // 2 * 3H*D

// ---------------- WMMA types ----------------
typedef __attribute__((ext_vector_type(16))) _Float16 v16h;
typedef __attribute__((ext_vector_type(8)))  _Float16 v8h;
typedef __attribute__((ext_vector_type(8)))  float    v8f;

// C(MxN,f32) = A(MxK,f16, optionally row-reversed) * W^T (W: NxK f16) + bias(N).
// One wave computes a 16x64 strip: A fragment reused by 4 WMMAs per K-step.
// A frag per lane: 16 f16 = two aligned v8h loads (K = h*16+0..7 and h*16+8..15).
// B frag per lane: 16 contiguous f16 of row (col+n), K = h*16+0..15 (one 32B load).
__global__ void __launch_bounds__(256)
wmma_gemm_bias(const _Float16* __restrict__ A, const _Float16* __restrict__ W,
               const float* __restrict__ bias, float* __restrict__ C,
               int M, int N, int Kd, int rev) {
  int lane = threadIdx.x & 31;
  int wave = threadIdx.x >> 5;
  int wpb  = blockDim.x >> 5;
  int wid  = blockIdx.x * wpb + wave;
  int cgrp = N >> 6;                       // 64-column groups
  int total = (M >> 4) * cgrp;
  if (wid >= total) return;
  int row0 = (wid / cgrp) << 4;
  int col0 = (wid % cgrp) << 6;

  int n = lane & 15;
  int h = lane >> 4;
  int row = row0 + n;
  if (rev) row = M - 1 - row;
  const _Float16* arow = A + (size_t)row * Kd;

  v8f acc[4] = {{}, {}, {}, {}};
  for (int k0 = 0; k0 < Kd; k0 += 32) {
    union UA { v16h v; v8h p[2]; } ua;
    ua.p[0] = *(const v8h*)(arow + k0 + h * 16);
    ua.p[1] = *(const v8h*)(arow + k0 + h * 16 + 8);
    v16h a = ua.v;
#pragma unroll
    for (int ct = 0; ct < 4; ++ct) {
      const _Float16* wrow = W + (size_t)(col0 + ct * 16 + n) * Kd + k0 + h * 16;
      v16h b = *(const v16h*)wrow;         // 32B aligned -> 2x global_load_b128
      acc[ct] = __builtin_amdgcn_wmma_f32_16x16x32_f16(false, a, false, b,
                                                       (short)0, acc[ct], false, false);
    }
  }
  int mb = h << 3;                          // C/D: lanes 0-15 rows 0..7, lanes 16-31 rows 8..15
#pragma unroll
  for (int ct = 0; ct < 4; ++ct) {
    float bb = bias[col0 + ct * 16 + n];
#pragma unroll
    for (int r = 0; r < 8; ++r)
      C[(size_t)(row0 + mb + r) * N + col0 + ct * 16 + n] = acc[ct][r] + bb;
  }
}

// ---------------- prep kernels ----------------
__global__ void gather_rows_h(const float* __restrict__ emb, const int* __restrict__ ids,
                              _Float16* __restrict__ dst, int rows, int width) {
  int i = blockIdx.x * blockDim.x + threadIdx.x;
  if (i >= rows * width) return;
  int r = i / width, j = i - r * width;
  dst[i] = (_Float16)emb[(size_t)ids[r] * width + j];
}

__global__ void f32_to_f16(const float* __restrict__ src, _Float16* __restrict__ dst, int n) {
  int i = blockIdx.x * blockDim.x + threadIdx.x;
  if (i < n) dst[i] = (_Float16)src[i];
}

__global__ void zero_fill(float* __restrict__ p, int n) {
  int i = blockIdx.x * blockDim.x + threadIdx.x;
  if (i < n) p[i] = 0.0f;
}

// aw_hh (HxH) -> transposed [j][r]
__global__ void transpose_mat(const float* __restrict__ src, float* __restrict__ dst,
                              int R, int Cw) {
  int i = blockIdx.x * blockDim.x + threadIdx.x;
  if (i >= R * Cw) return;
  int r = i / Cw, j = i - r * Cw;
  dst[(size_t)j * R + r] = src[i];
}

// gate-pack: src (3H x H) row-major -> dst float4[j][r] = {W_i[r][j], W_o[r][j], W_g[r][j], 0}
__global__ void pack_gates(const float* __restrict__ src, float* __restrict__ dst) {
  int i = blockIdx.x * blockDim.x + threadIdx.x;
  if (i >= HH * HH) return;
  int j = i / HH, r = i - j * HH;
  float4 v;
  v.x = src[(size_t)(0 * HH + r) * HH + j];
  v.y = src[(size_t)(1 * HH + r) * HH + j];
  v.z = src[(size_t)(2 * HH + r) * HH + j];
  v.w = 0.0f;
  ((float4*)dst)[(size_t)j * HH + r] = v;
}

// ---------------- sequential lattice scan ----------------
__device__ inline float sigm(float x) { return 1.0f / (1.0f + __expf(-x)); }

__global__ void __launch_bounds__(HH)
lattice_scan(float* __restrict__ ws,
             const int* __restrict__ eidx0, const int* __restrict__ eidx1,
             const int* __restrict__ sidx0, const int* __restrict__ sidx1,
             float* __restrict__ out) {
  const int dir = blockIdx.x;   // 0 = forward, 1 = backward
  const int tid = threadIdx.x;  // 0..255

  const float*  char_pre  = ws + OFF_CHAR  + (size_t)dir * TT * H3;
  const float*  alpha_pre = ws + OFF_ALPHA + (size_t)dir * TT * HH;
  const float*  word_pre  = ws + OFF_WORD  + (size_t)dir * (GG + 1) * H3;
  float*        buf       = ws + OFF_BUF   + (size_t)dir * (GG + 1) * HH;
  const float*  awT       = ws + OFF_AWT   + (size_t)dir * HH * HH;
  const float4* whh4      = (const float4*)(ws + OFF_WHH4)  + (size_t)dir * HH * HH;
  const float4* wwhh4     = (const float4*)(ws + OFF_WWHH4) + (size_t)dir * HH * HH;
  const int*    eidx      = dir ? eidx1 : eidx0;
  const int*    sidx      = dir ? sidx1 : sidx0;

  __shared__ float h_sh[HH];
  __shared__ float cin_sh[KK][HH];

  float c = 0.0f;
  h_sh[tid] = 0.0f;
  __syncthreads();

  for (int t = 0; t < TT; ++t) {
    const float* cp = char_pre + (size_t)t * H3;
    if (t + 1 < TT) {  // global_prefetch of next step's precomputed rows
      __builtin_prefetch(char_pre + (size_t)(t + 1) * H3 + tid, 0, 0);
      __builtin_prefetch(alpha_pre + (size_t)(t + 1) * HH + tid, 0, 0);
    }

    // gates = char_pre[t] + w_hh @ h   (one b128 load per j: packed {i,o,g})
    float accI = cp[tid], accO = cp[HH + tid], accG = cp[2 * HH + tid];
#pragma unroll 4
    for (int j = 0; j < HH; ++j) {
      float hv = h_sh[j];
      float4 w = whh4[(size_t)j * HH + tid];
      accI += w.x * hv;
      accO += w.y * hv;
      accG += w.z * hv;
    }
    float ig = sigm(accI), og = sigm(accO), gg = tanhf(accG);

    int e[KK];
    bool anyv = false;
#pragma unroll
    for (int k = 0; k < KK; ++k) {
      e[k] = eidx[t * KK + k];
      anyv = anyv || (e[k] < GG);
      cin_sh[k][tid] = buf[(size_t)e[k] * HH + tid];
    }
    __syncthreads();  // cin_sh fully written

    // alpha[k] = sigmoid(alpha_pre[t] + c_in[k] @ aw_hh^T)
    float aacc[KK];
    float ap = alpha_pre[(size_t)t * HH + tid];
#pragma unroll
    for (int k = 0; k < KK; ++k) aacc[k] = ap;
#pragma unroll 4
    for (int j = 0; j < HH; ++j) {
      float w = awT[(size_t)j * HH + tid];
#pragma unroll
      for (int k = 0; k < KK; ++k) aacc[k] += w * cin_sh[k][j];
    }

    float ei  = __expf(ig);
    float num = ei * gg;
    float den = ei;
#pragma unroll
    for (int k = 0; k < KK; ++k) {
      if (e[k] < GG) {
        float ew = __expf(sigm(aacc[k]));
        num += ew * cin_sh[k][tid];
        den += ew;
      }
    }
    float c_soft = num / den;
    float c_cpl  = (1.0f - ig) * c + ig * gg;
    float c_new  = anyv ? c_soft : c_cpl;
    float h_new  = og * tanhf(c_new);

    h_sh[tid] = h_new;   // all reads of old h_sh completed before previous sync
    __syncthreads();     // h_new visible to all lanes

    // word cells: wg = word_pre[sidx] + ww_hh @ h_new
    float wI = 0.0f, wF = 0.0f, wG = 0.0f;
#pragma unroll 4
    for (int j = 0; j < HH; ++j) {
      float hv = h_sh[j];
      float4 w = wwhh4[(size_t)j * HH + tid];
      wI += w.x * hv;
      wF += w.y * hv;
      wG += w.z * hv;
    }
#pragma unroll
    for (int k = 0; k < KK; ++k) {
      int s = sidx[t * KK + k];
      const float* wp = word_pre + (size_t)s * H3;
      float iw = sigm(wp[tid] + wI);
      float fw = sigm(wp[HH + tid] + wF);
      float gw = tanhf(wp[2 * HH + tid] + wG);
      buf[(size_t)s * HH + tid] = fw * c_new + iw * gw;
    }

    if (dir == 0) out[(size_t)t * (2 * HH) + tid] = h_new;
    else          out[(size_t)(TT - 1 - t) * (2 * HH) + HH + tid] = h_new;

    c = c_new;
    __syncthreads();  // buf stores visible before next step's gather
  }
}

// ---------------- launcher ----------------
extern "C" void kernel_launch(void* const* d_in, const int* in_sizes, int n_in,
                              void* d_out, int out_size, void* d_ws, size_t ws_size,
                              hipStream_t stream) {
  (void)in_sizes; (void)n_in; (void)out_size; (void)ws_size;
  const int*   tokens    = (const int*)d_in[0];
  const int*   gaz_ids   = (const int*)d_in[1];
  const int*   sidx_f    = (const int*)d_in[2];
  const int*   eidx_f    = (const int*)d_in[3];
  const int*   sidx_b    = (const int*)d_in[4];
  const int*   eidx_b    = (const int*)d_in[5];
  const float* token_emb = (const float*)d_in[6];
  const float* gaz_emb   = (const float*)d_in[7];
  const float* w_ih[2]  = { (const float*)d_in[8],  (const float*)d_in[17] };
  const float* w_hh[2]  = { (const float*)d_in[9],  (const float*)d_in[18] };
  const float* b_[2]    = { (const float*)d_in[10], (const float*)d_in[19] };
  const float* aw_ih[2] = { (const float*)d_in[11], (const float*)d_in[20] };
  const float* aw_hh[2] = { (const float*)d_in[12], (const float*)d_in[21] };
  const float* ab_[2]   = { (const float*)d_in[13], (const float*)d_in[22] };
  const float* ww_ih[2] = { (const float*)d_in[14], (const float*)d_in[23] };
  const float* ww_hh[2] = { (const float*)d_in[15], (const float*)d_in[24] };
  const float* wb_[2]   = { (const float*)d_in[16], (const float*)d_in[25] };

  float*     ws  = (float*)d_ws;
  _Float16*  wsh = (_Float16*)(ws + F32_TOTAL);
  float*     out = (float*)d_out;

  const int B = 256;
  auto blocks = [](long n, int b) { return (int)((n + b - 1) / b); };

  // 1) gathers straight into f16
  gather_rows_h<<<blocks((long)TT * DD, B), B, 0, stream>>>(token_emb, tokens, wsh + HOFF_X, TT, DD);
  gather_rows_h<<<blocks((long)GG * DD, B), B, 0, stream>>>(gaz_emb, gaz_ids, wsh + HOFF_GAZ, GG, DD);

  // 2) zero buf region + word_pre sentinel rows
  zero_fill<<<blocks((long)2 * (GG + 1) * HH, B), B, 0, stream>>>(ws + OFF_BUF, 2 * (GG + 1) * HH);
  zero_fill<<<blocks(H3, B), B, 0, stream>>>(ws + OFF_WORD + (size_t)GG * H3, H3);
  zero_fill<<<blocks(H3, B), B, 0, stream>>>(ws + OFF_WORD + (size_t)(GG + 1) * H3 + (size_t)GG * H3, H3);

  // 3) weight prep: f16 copies of *_ih, packed/transposed recurrent weights
  for (int d = 0; d < 2; ++d) {
    f32_to_f16<<<blocks((long)H3 * DD, B), B, 0, stream>>>(w_ih[d],  wsh + HOFF_WIH  + (size_t)d * H3 * DD, H3 * DD);
    f32_to_f16<<<blocks((long)HH * DD, B), B, 0, stream>>>(aw_ih[d], wsh + HOFF_AWIH + (size_t)d * HH * DD, HH * DD);
    f32_to_f16<<<blocks((long)H3 * DD, B), B, 0, stream>>>(ww_ih[d], wsh + HOFF_WWIH + (size_t)d * H3 * DD, H3 * DD);
    transpose_mat<<<blocks((long)HH * HH, B), B, 0, stream>>>(aw_hh[d], ws + OFF_AWT + (size_t)d * HH * HH, HH, HH);
    pack_gates<<<blocks((long)HH * HH, B), B, 0, stream>>>(w_hh[d],  ws + OFF_WHH4  + (size_t)d * HH * HH * 4);
    pack_gates<<<blocks((long)HH * HH, B), B, 0, stream>>>(ww_hh[d], ws + OFF_WWHH4 + (size_t)d * HH * HH * 4);
  }

  // 4) WMMA pre-GEMMs: one wave per 16x64 strip (4 wmma per K-step), 8 waves/block
  auto gemm = [&](const _Float16* A, const _Float16* W, const float* bias, float* C,
                  int M, int N, int Kd, int rev) {
    int waves = (M / 16) * (N / 64);
    wmma_gemm_bias<<<blocks(waves, 8), 256, 0, stream>>>(A, W, bias, C, M, N, Kd, rev);
  };
  for (int d = 0; d < 2; ++d) {
    gemm(wsh + HOFF_X,   wsh + HOFF_WIH  + (size_t)d * H3 * DD, b_[d],
         ws + OFF_CHAR  + (size_t)d * TT * H3,       TT, H3, DD, d);
    gemm(wsh + HOFF_X,   wsh + HOFF_AWIH + (size_t)d * HH * DD, ab_[d],
         ws + OFF_ALPHA + (size_t)d * TT * HH,       TT, HH, DD, d);
    gemm(wsh + HOFF_GAZ, wsh + HOFF_WWIH + (size_t)d * H3 * DD, wb_[d],
         ws + OFF_WORD  + (size_t)d * (GG + 1) * H3, GG, H3, DD, 0);
  }

  // 5) sequential scan: one block per direction
  lattice_scan<<<2, HH, 0, stream>>>(ws, eidx_f, eidx_b, sidx_f, sidx_b, out);
}
